// AdaptiveLinearWithChannel_36481452212285
// MI455X (gfx1250) — matrix operand order repaired and checked by reading, not compiled, not run on hardware
//
#include <hip/hip_runtime.h>

#define NSEL   192
#define NPTS   2048
#define CIN    256
#define COUT   256
#define NTILES 8

#define BM  64
#define BN  256
#define KC  32
#define LDA (KC + 2)   // padded row stride (ushorts) -> 68B, breaks 4-way bank conflicts
#define LDB (KC + 2)

typedef __attribute__((ext_vector_type(16))) __bf16 bf16x16;
typedef __attribute__((ext_vector_type(8)))  float  f32x8;

union Frag {
    unsigned int u[8];
    bf16x16      v;
};

// round-to-nearest-even f32 -> bf16, two at a time, packed into one dword
__device__ __forceinline__ unsigned int pack2_bf16(float a, float b) {
    unsigned int ua = __float_as_uint(a);
    unsigned int ub = __float_as_uint(b);
    ua += 0x7fffu + ((ua >> 16) & 1u);
    ub += 0x7fffu + ((ub >> 16) & 1u);
    return (ua >> 16) | (ub & 0xffff0000u);
}

__device__ __forceinline__ unsigned short to_bf16(float a) {
    unsigned int ua = __float_as_uint(a);
    ua += 0x7fffu + ((ua >> 16) & 1u);
    return (unsigned short)(ua >> 16);
}

__global__ __launch_bounds__(256) void adaptive_linear_wmma(
    const float* __restrict__ x,        // [NSEL, NPTS, CIN]
    const float* __restrict__ w,        // [NCHAN, NTILES, CIN, COUT]
    const float* __restrict__ bias,     // [NCHAN, NTILES, 1, COUT]
    const int*   __restrict__ indices,  // [NSEL]
    const int*   __restrict__ tptr,     // [1]
    float*       __restrict__ out)      // [NSEL, NPTS, COUT]
{
    const int msel = blockIdx.y;   // selected channel 0..191
    const int mblk = blockIdx.x;   // row-tile 0..31
    const int t    = tptr[0];
    const int chan = indices[msel];

    const float* xg = x   + (size_t)msel * NPTS * CIN + (size_t)mblk * BM * CIN;
    const float* wg = w   + ((size_t)chan * NTILES + t) * (size_t)(CIN * COUT);
    const float* bg = bias + ((size_t)chan * NTILES + t) * COUT;
    float*       og = out + (size_t)msel * NPTS * COUT + (size_t)mblk * BM * COUT;

    __shared__ unsigned short As[BM][LDA];   // A tile, row-major (m, k), bf16
    __shared__ unsigned short Bs[BN][LDB];   // B tile transposed (n, k), bf16

    const int tid  = threadIdx.x;
    const int lane = tid & 31;
    const int wave = tid >> 5;
    const int l    = lane & 15;   // N / M position within 16
    const int hi   = lane >> 4;   // half-wave select
    const int wm   = wave >> 2;   // wave M index: 0..1
    const int wn   = wave & 3;    // wave N index: 0..3

    f32x8 acc[2][4] = {};

    for (int kc = 0; kc < CIN; kc += KC) {
        // ---- stage A tile: 64 rows x 32 k (f32 -> bf16) ----
        #pragma unroll
        for (int it = 0; it < 2; ++it) {
            int row = (tid >> 3) + it * 32;
            int q   = tid & 7;
            float4 v = *reinterpret_cast<const float4*>(xg + (size_t)row * CIN + kc + q * 4);
            unsigned int* dst = reinterpret_cast<unsigned int*>(&As[row][q * 4]);
            dst[0] = pack2_bf16(v.x, v.y);
            dst[1] = pack2_bf16(v.z, v.w);
        }
        // ---- stage B tile: 32 k-rows x 256 n (coalesced read, transposed store) ----
        #pragma unroll
        for (int it = 0; it < 8; ++it) {
            int gid = tid + it * 256;
            int k   = gid >> 6;
            int n0  = (gid & 63) * 4;
            float4 v = *reinterpret_cast<const float4*>(wg + (size_t)(kc + k) * COUT + n0);
            Bs[n0 + 0][k] = to_bf16(v.x);
            Bs[n0 + 1][k] = to_bf16(v.y);
            Bs[n0 + 2][k] = to_bf16(v.z);
            Bs[n0 + 3][k] = to_bf16(v.w);
        }
        __syncthreads();

        // ---- A fragments (ISA 16-bit A 16x32 layout) ----
        Frag afr[2];
        #pragma unroll
        for (int mi = 0; mi < 2; ++mi) {
            int row = wm * 32 + mi * 16 + l;
            #pragma unroll
            for (int v = 0; v < 8; ++v) {
                // hi=0: v<4 -> K=2v ; v>=4 -> K=2v+8 (16..23). hi=1: +8.
                int kp = 2 * v + (v >= 4 ? 8 : 0) + hi * 8;
                afr[mi].u[v] = *reinterpret_cast<const unsigned int*>(&As[row][kp]);
            }
        }

        // ---- B fragments + WMMA ----
        #pragma unroll
        for (int nj = 0; nj < 4; ++nj) {
            int col = wn * 64 + nj * 16 + l;
            Frag bfr;
            #pragma unroll
            for (int v = 0; v < 8; ++v) {
                // hi=0: v<4 -> K=2v ; v>=4 -> K=2v+8 (16..23). hi=1: +8.
                int kp = 2 * v + (v >= 4 ? 8 : 0) + hi * 8;
                bfr.u[v] = *reinterpret_cast<const unsigned int*>(&Bs[col][kp]);
            }
            #pragma unroll
            for (int mi = 0; mi < 2; ++mi) {
                acc[mi][nj] = __builtin_amdgcn_wmma_f32_16x16x32_bf16(
                    false, afr[mi].v, false, bfr.v,
                    (short)0, acc[mi][nj], false, false);
            }
        }
        __syncthreads();
    }

    // ---- epilogue: bias add + f32 store (C/D layout: VGPR r -> M=r+8*hi, N=l) ----
    #pragma unroll
    for (int nj = 0; nj < 4; ++nj) {
        int col = wn * 64 + nj * 16 + l;
        float bv = bg[col];
        #pragma unroll
        for (int mi = 0; mi < 2; ++mi) {
            #pragma unroll
            for (int r = 0; r < 8; ++r) {
                int row = wm * 32 + mi * 16 + r + 8 * hi;
                og[(size_t)row * COUT + col] = acc[mi][nj][r] + bv;
            }
        }
    }
}

extern "C" void kernel_launch(void* const* d_in, const int* in_sizes, int n_in,
                              void* d_out, int out_size, void* d_ws, size_t ws_size,
                              hipStream_t stream) {
    const float* x   = (const float*)d_in[0];
    const float* w   = (const float*)d_in[1];
    const float* b   = (const float*)d_in[2];
    const int*   idx = (const int*)d_in[3];
    const int*   t   = (const int*)d_in[4];
    float* out = (float*)d_out;

    dim3 grid(NPTS / BM, NSEL);   // 32 x 192 = 6144 workgroups
    adaptive_linear_wmma<<<grid, 256, 0, stream>>>(x, w, b, idx, t, out);
}
